// MatchLayer_70205535421126
// MI455X (gfx1250) — 8-bit matrix operand order repaired and checked by reading, compiled, not run on hardware
//
#include <hip/hip_runtime.h>
#include <hip/hip_bf16.h>

typedef int v8i __attribute__((ext_vector_type(8)));

#define Bn 16384
#define Fn 4096
#define Kp 256
#define Pp 16

#define ROWS_PER_WAVE 16
#define WAVES_PER_BLOCK 8
#define LDS_STRIDE 272

__global__ void zero_S_kernel(unsigned int* __restrict__ ws, int n_words) {
    int i = blockIdx.x * blockDim.x + threadIdx.x;
    if (i < n_words) ws[i] = 0u;
}

__global__ void build_S_kernel(const int* __restrict__ idx,
                               unsigned char* __restrict__ S) {
    int k = blockIdx.x * blockDim.x + threadIdx.x;
    if (k >= Kp) return;
    for (int p = 0; p < Pp; ++p) {
        int f = idx[k * Pp + p];
        S[(size_t)f * Kp + k] = (unsigned char)(S[(size_t)f * Kp + k] + 1);
    }
}

__device__ __forceinline__ unsigned pack4(float4 f) {
    return (f.x > 0.5f ? 1u : 0u)
         | ((f.y > 0.5f ? 1u : 0u) << 8)
         | ((f.z > 0.5f ? 1u : 0u) << 16)
         | ((f.w > 0.5f ? 1u : 0u) << 24);
}

__global__ __launch_bounds__(256, 1)
void match_wmma_kernel(const float* __restrict__ in,
                       const unsigned char* __restrict__ S,
                       float* __restrict__ out) {
    __shared__ unsigned char sB[2][64 * LDS_STRIDE];

    const int tid  = threadIdx.x;
    const int lane = tid & 31;
    const int wave = tid >> 5;
    const int rowBase = (blockIdx.x * WAVES_PER_BLOCK + wave) * ROWS_PER_WAVE;

    const int   llo    = lane & 15;
    const int   khalf  = (lane >> 4) * 8;
    const float* rowPtr = in + (size_t)(rowBase + llo) * Fn;

    v8i acc[16];
#pragma unroll
    for (int nt = 0; nt < 16; ++nt)
        acc[nt] = (v8i){0, 0, 0, 0, 0, 0, 0, 0};

    uint4 pf[4];
#pragma unroll
    for (int c = 0; c < 4; ++c) {
        int chunk = tid + c * 256;
        int r   = chunk >> 4;
        int col = (chunk & 15) * 16;
        pf[c] = *(const uint4*)(S + (size_t)r * Kp + col);
    }

    for (int ks = 0; ks < Fn / 64; ++ks) {
        unsigned char* buf = sB[ks & 1];

#pragma unroll
        for (int c = 0; c < 4; ++c) {
            int chunk = tid + c * 256;
            int r   = chunk >> 4;
            int col = (chunk & 15) * 16;
            *(uint4*)(buf + r * LDS_STRIDE + col) = pf[c];
        }
        if (ks + 1 < Fn / 64) {
            int kb = (ks + 1) * 64;
#pragma unroll
            for (int c = 0; c < 4; ++c) {
                int chunk = tid + c * 256;
                int r   = chunk >> 4;
                int col = (chunk & 15) * 16;
                pf[c] = *(const uint4*)(S + (size_t)(kb + r) * Kp + col);
            }
        }
        __syncthreads();

        const int kbase = ks * 64;
        v8i a;
#pragma unroll
        for (int g = 0; g < 4; ++g) {
            float4 f0 = *(const float4*)(rowPtr + kbase + g * 16 + khalf);
            float4 f1 = *(const float4*)(rowPtr + kbase + g * 16 + khalf + 4);
            a[2 * g]     = (int)pack4(f0);
            a[2 * g + 1] = (int)pack4(f1);
        }

        const unsigned char* r0 = buf + llo;
#pragma unroll
        for (int nt = 0; nt < 16; ++nt) {
            v8i b;
#pragma unroll
            for (int w = 0; w < 8; ++w) {
                unsigned v = 0u;
#pragma unroll
                for (int q = 0; q < 4; ++q) {
                    int kk = 16 * (w >> 1) + khalf + 4 * (w & 1) + q;
                    v |= (unsigned)r0[kk * LDS_STRIDE + nt * 16] << (8 * q);
                }
                b[w] = (int)v;
            }
            acc[nt] = __builtin_amdgcn_wmma_i32_16x16x64_iu8(
                false, a, false, b, acc[nt], false, false);
        }
    }

#pragma unroll
    for (int j = 0; j < 8; ++j) {
        int m = 0;
#pragma unroll
        for (int nt = 0; nt < 16; ++nt)
            m |= (acc[nt][j] == Pp) ? 1 : 0;
        unsigned msk = __builtin_amdgcn_ballot_w32(m != 0);
        if (lane == 0)
            out[rowBase + j]     = (msk & 0xFFFFu) ? 1.0f : 0.0f;
        if (lane == 16)
            out[rowBase + 8 + j] = (msk >> 16)     ? 1.0f : 0.0f;
    }
}

extern "C" void kernel_launch(void* const* d_in, const int* in_sizes, int n_in,
                              void* d_out, int out_size, void* d_ws, size_t ws_size,
                              hipStream_t stream) {
    const float* in  = (const float*)d_in[0];
    const int*   idx = (const int*)d_in[1];
    float*       out = (float*)d_out;
    unsigned char* S = (unsigned char*)d_ws;

    const int n_words = (Fn * Kp) / 4;
    zero_S_kernel<<<(n_words + 255) / 256, 256, 0, stream>>>((unsigned int*)S, n_words);
    build_S_kernel<<<1, 256, 0, stream>>>(idx, S);

    const int grid = Bn / (ROWS_PER_WAVE * WAVES_PER_BLOCK);
    match_wmma_kernel<<<grid, 256, 0, stream>>>(in, S, out);
}
